// SwitchSparseMoe_13125420057241
// MI455X (gfx1250) — matrix operand order repaired and checked by reading, not compiled, not run on hardware
//
#include <hip/hip_runtime.h>
#include <hip/hip_bf16.h>

#define B_  2
#define S_  4096
#define D_  1024
#define FF_ 4096
#define E_  8
#define C_  1024

typedef __attribute__((ext_vector_type(16))) __bf16 bf16x16;
typedef __attribute__((ext_vector_type(8)))  __bf16 bf16x8;
typedef __attribute__((ext_vector_type(8)))  float  f32x8;
typedef __attribute__((ext_vector_type(4)))  int    int4v;

#define GLOBAL_AS __attribute__((address_space(1)))
#define LOCAL_AS  __attribute__((address_space(3)))

// 16B global -> LDS async copy (gfx1250 GLOBAL_LOAD_ASYNC_TO_LDS_B128).
__device__ inline void async_copy16(const void* gsrc, void* ldst) {
#if __has_builtin(__builtin_amdgcn_global_load_async_to_lds_b128)
  __builtin_amdgcn_global_load_async_to_lds_b128(
      (GLOBAL_AS int4v*)gsrc, (LOCAL_AS int4v*)ldst, 0, 0);
#else
  *(bf16x8*)ldst = *(const bf16x8*)gsrc;
#endif
}

__device__ inline void async_wait0() {
#if __has_builtin(__builtin_amdgcn_s_wait_asynccnt)
  __builtin_amdgcn_s_wait_asynccnt(0);
#else
  asm volatile("s_wait_asynccnt 0x0" ::: "memory");
#endif
}

__device__ inline f32x8 wmma_bf16(bf16x16 a, bf16x16 b, f32x8 c) {
  // D = A(16x32 bf16) * B(32x16 bf16) + C(16x16 f32)
  return __builtin_amdgcn_wmma_f32_16x16x32_bf16(false, a, false, b,
                                                 (short)0, c, false, false);
}

__device__ inline bf16x16 frag_cat(bf16x8 lo, bf16x8 hi) {
  return __builtin_shufflevector(lo, hi, 0, 1, 2, 3, 4, 5, 6, 7,
                                 8, 9, 10, 11, 12, 13, 14, 15);
}

// A tile in LDS: [128 rows][32 k] bf16, row-major.
__device__ inline bf16x16 load_frag_a(const __bf16* lds, int row, int h) {
  const bf16x8* p = (const bf16x8*)(lds + row * 32 + 8 * h);
  const bf16x8* q = (const bf16x8*)(lds + row * 32 + 16 + 8 * h);
  return frag_cat(*p, *q);
}

// B tile in LDS: [128 n][32 k] bf16 (N-major / transposed).
__device__ inline bf16x16 load_frag_b(const __bf16* lds, int n, int h) {
  const bf16x8* p = (const bf16x8*)(lds + n * 32 + 8 * h);
  const bf16x8* q = (const bf16x8*)(lds + n * 32 + 16 + 8 * h);
  return frag_cat(*p, *q);
}

// ---------------------------------------------------------------- router ----
__global__ __launch_bounds__(256) void router_kernel(
    const float* __restrict__ hs, const float* __restrict__ gw,
    float* __restrict__ logits_out, int* __restrict__ sel,
    float* __restrict__ maxp) {
  const int wave = threadIdx.x >> 5;
  const int lane = threadIdx.x & 31;
  const int token = blockIdx.x * 8 + wave;  // [0, B*S)
  const float* x = hs + (size_t)token * D_;

  float acc[E_] = {};
  for (int d = lane; d < D_; d += 32) {
    const float h = x[d];
#pragma unroll
    for (int e = 0; e < E_; ++e) acc[e] += h * gw[d * E_ + e];
  }
#pragma unroll
  for (int e = 0; e < E_; ++e)
    for (int off = 16; off > 0; off >>= 1)
      acc[e] += __shfl_down(acc[e], off, 32);

  if (lane == 0) {
    float mx = acc[0];
    int am = 0;
#pragma unroll
    for (int e = 1; e < E_; ++e)
      if (acc[e] > mx) { mx = acc[e]; am = e; }
    float se = 0.0f;
#pragma unroll
    for (int e = 0; e < E_; ++e) {
      se += __expf(acc[e] - mx);
      logits_out[(size_t)token * E_ + e] = acc[e];
    }
    sel[token] = am;
    maxp[token] = __builtin_amdgcn_rcpf(se);  // max softmax prob = 1/sum
  }
}

// ------------------------------------------------------------ inv init ------
__global__ __launch_bounds__(256) void inv_init_kernel(int* __restrict__ inv) {
  inv[blockIdx.x * 256 + threadIdx.x] = -1;
}

// --------------------------------- capacity scan / inverse map (per batch) --
__global__ __launch_bounds__(256) void priority_kernel(
    const int* __restrict__ sel, int* __restrict__ inv) {
  __shared__ int cnt[256][E_];
  const int b = blockIdx.x;
  const int t = threadIdx.x;
  const int TPT = S_ / 256;  // 16 tokens per thread
  const int s0 = t * TPT;

  int local[E_] = {};
#pragma unroll
  for (int i = 0; i < TPT; ++i) local[sel[b * S_ + s0 + i]]++;
#pragma unroll
  for (int e = 0; e < E_; ++e) cnt[t][e] = local[e];
  __syncthreads();

  // exclusive prefix over threads, one expert per scanning thread
  if (t < E_) {
    int run = 0;
    for (int i = 0; i < 256; ++i) {
      const int c = cnt[i][t];
      cnt[i][t] = run;
      run += c;
    }
  }
  __syncthreads();

  int pos[E_];
#pragma unroll
  for (int e = 0; e < E_; ++e) pos[e] = cnt[t][e];
#pragma unroll
  for (int i = 0; i < TPT; ++i) {
    const int s = s0 + i;
    const int e = sel[b * S_ + s];
    const int p = pos[e]++;
    if (p < C_) inv[((size_t)b * E_ + e) * C_ + p] = s;
  }
}

// ----------------------------------------------------------- dispatch -------
__global__ __launch_bounds__(256) void dispatch_kernel(
    const float* __restrict__ hs, const int* __restrict__ inv,
    __bf16* __restrict__ X) {
  const int row = blockIdx.x;  // [0, B*E*C)
  const int b = row / (E_ * C_);
  const int s = inv[row];
  __bf16* dst = X + (size_t)row * D_;
  const int i = threadIdx.x * 4;
  if (s >= 0) {
    const float4 v = *(const float4*)(hs + ((size_t)b * S_ + s) * D_ + i);
    dst[i + 0] = (__bf16)v.x;
    dst[i + 1] = (__bf16)v.y;
    dst[i + 2] = (__bf16)v.z;
    dst[i + 3] = (__bf16)v.w;
  } else {
    dst[i + 0] = (__bf16)0.0f;
    dst[i + 1] = (__bf16)0.0f;
    dst[i + 2] = (__bf16)0.0f;
    dst[i + 3] = (__bf16)0.0f;
  }
}

// ------------------------------------------- GEMM1: H = silu(XW1)*(XW3) -----
__global__ __launch_bounds__(256) void moe_gemm1_kernel(
    const __bf16* __restrict__ X, const float* __restrict__ w1,
    const float* __restrict__ w3, __bf16* __restrict__ H) {
  __shared__ __align__(16) __bf16 lA[128 * 32];
  __shared__ __align__(16) __bf16 lB1[128 * 32];
  __shared__ __align__(16) __bf16 lB3[128 * 32];

  const int nT = blockIdx.x % (FF_ / 128);
  const int mT = (blockIdx.x / (FF_ / 128)) % (C_ / 128);
  const int be = blockIdx.x / ((FF_ / 128) * (C_ / 128));
  const int e = be % E_;

  const __bf16* A = X + (size_t)be * C_ * D_ + (size_t)mT * 128 * D_;
  const float* B1 = w1 + (size_t)e * D_ * FF_ + (size_t)nT * 128;
  const float* B3 = w3 + (size_t)e * D_ * FF_ + (size_t)nT * 128;
  __bf16* Ho = H + (size_t)be * C_ * FF_ + (size_t)mT * 128 * FF_ +
               (size_t)nT * 128;

  const int tid = threadIdx.x;
  const int lane = tid & 31;
  const int wave = tid >> 5;
  const int wm = wave & 1;   // 2 waves in M: 64 rows each
  const int wn = wave >> 1;  // 4 waves in N: 32 cols each
  const int lm = lane & 15;
  const int h = lane >> 4;

  f32x8 accG[4][2] = {};
  f32x8 accU[4][2] = {};

  for (int it = 0; it < D_ / 32; ++it) {
    const int k0 = it * 32;
    // stage A: 128x32 bf16 row-major, 32B/thread via async LDS copies
    {
      const int row = tid >> 1;
      const int col = (tid & 1) * 16;
      const __bf16* src = A + (size_t)row * D_ + k0 + col;
      __bf16* dst = lA + row * 32 + col;
      async_copy16(src, dst);
      async_copy16(src + 8, dst + 8);
    }
    // stage B: 32x128 fp32 -> bf16, transposed to N-major LDS
    {
      const int k = tid >> 3;
      const int n0 = (tid & 7) * 16;
      const float* s1 = B1 + (size_t)(k0 + k) * FF_ + n0;
      const float* s3 = B3 + (size_t)(k0 + k) * FF_ + n0;
      __builtin_prefetch(s1 + (size_t)32 * FF_, 0, 1);  // next k-tile
      __builtin_prefetch(s3 + (size_t)32 * FF_, 0, 1);
#pragma unroll
      for (int i = 0; i < 16; ++i) {
        lB1[(n0 + i) * 32 + k] = (__bf16)s1[i];
        lB3[(n0 + i) * 32 + k] = (__bf16)s3[i];
      }
    }
    async_wait0();
    __syncthreads();

    bf16x16 af[4];
#pragma unroll
    for (int mi = 0; mi < 4; ++mi)
      af[mi] = load_frag_a(lA, wm * 64 + mi * 16 + lm, h);
    bf16x16 b1f[2], b3f[2];
#pragma unroll
    for (int ni = 0; ni < 2; ++ni) {
      b1f[ni] = load_frag_b(lB1, wn * 32 + ni * 16 + lm, h);
      b3f[ni] = load_frag_b(lB3, wn * 32 + ni * 16 + lm, h);
    }
#pragma unroll
    for (int mi = 0; mi < 4; ++mi)
#pragma unroll
      for (int ni = 0; ni < 2; ++ni) {
        accG[mi][ni] = wmma_bf16(af[mi], b1f[ni], accG[mi][ni]);
        accU[mi][ni] = wmma_bf16(af[mi], b3f[ni], accU[mi][ni]);
      }
    __syncthreads();
  }

#pragma unroll
  for (int mi = 0; mi < 4; ++mi)
#pragma unroll
    for (int ni = 0; ni < 2; ++ni)
#pragma unroll
      for (int r = 0; r < 8; ++r) {
        const float g = accG[mi][ni][r];
        const float u = accU[mi][ni][r];
        // silu(g)*u with fast v_rcp sigmoid
        const float v = g * __builtin_amdgcn_rcpf(1.0f + __expf(-g)) * u;
        const int m = wm * 64 + mi * 16 + r + 8 * h;
        const int n = wn * 32 + ni * 16 + lm;
        Ho[(size_t)m * FF_ + n] = (__bf16)v;
      }
}

// ------------------------------------- fill: out = maxp * hidden (all) ------
__global__ __launch_bounds__(256) void fill_kernel(
    const float* __restrict__ hs, const float* __restrict__ maxp,
    float* __restrict__ out) {
  const size_t idx = ((size_t)blockIdx.x * 256 + threadIdx.x) * 4;
  const float sc = maxp[idx / D_];
  const float4 v = *(const float4*)(hs + idx);
  float4 r;
  r.x = sc * v.x;
  r.y = sc * v.y;
  r.z = sc * v.z;
  r.w = sc * v.w;
  *(float4*)(out + idx) = r;
}

// ------------------- GEMM2: Y = H*W2, scatter rows to out via inv, *maxp ----
__global__ __launch_bounds__(256) void moe_gemm2_kernel(
    const __bf16* __restrict__ H, const float* __restrict__ w2,
    const int* __restrict__ inv, const float* __restrict__ maxp,
    float* __restrict__ out) {
  __shared__ __align__(16) __bf16 lA[128 * 32];
  __shared__ __align__(16) __bf16 lB[128 * 32];

  const int nT = blockIdx.x % (D_ / 128);
  const int mT = (blockIdx.x / (D_ / 128)) % (C_ / 128);
  const int be = blockIdx.x / ((D_ / 128) * (C_ / 128));
  const int e = be % E_;
  const int b = be / E_;

  const __bf16* A = H + (size_t)be * C_ * FF_ + (size_t)mT * 128 * FF_;
  const float* Bw = w2 + (size_t)e * FF_ * D_ + (size_t)nT * 128;

  const int tid = threadIdx.x;
  const int lane = tid & 31;
  const int wave = tid >> 5;
  const int wm = wave & 1;
  const int wn = wave >> 1;
  const int lm = lane & 15;
  const int h = lane >> 4;

  f32x8 acc[4][2] = {};

  for (int it = 0; it < FF_ / 32; ++it) {
    const int k0 = it * 32;
    {
      const int row = tid >> 1;
      const int col = (tid & 1) * 16;
      const __bf16* src = A + (size_t)row * FF_ + k0 + col;
      __bf16* dst = lA + row * 32 + col;
      async_copy16(src, dst);
      async_copy16(src + 8, dst + 8);
    }
    {
      const int k = tid >> 3;
      const int n0 = (tid & 7) * 16;
      const float* s2 = Bw + (size_t)(k0 + k) * D_ + n0;
      __builtin_prefetch(s2 + (size_t)32 * D_, 0, 1);
#pragma unroll
      for (int i = 0; i < 16; ++i) lB[(n0 + i) * 32 + k] = (__bf16)s2[i];
    }
    async_wait0();
    __syncthreads();

    bf16x16 af[4];
#pragma unroll
    for (int mi = 0; mi < 4; ++mi)
      af[mi] = load_frag_a(lA, wm * 64 + mi * 16 + lm, h);
    bf16x16 bf[2];
#pragma unroll
    for (int ni = 0; ni < 2; ++ni)
      bf[ni] = load_frag_b(lB, wn * 32 + ni * 16 + lm, h);
#pragma unroll
    for (int mi = 0; mi < 4; ++mi)
#pragma unroll
      for (int ni = 0; ni < 2; ++ni)
        acc[mi][ni] = wmma_bf16(af[mi], bf[ni], acc[mi][ni]);
    __syncthreads();
  }

#pragma unroll
  for (int mi = 0; mi < 4; ++mi)
#pragma unroll
    for (int r = 0; r < 8; ++r) {
      const int m = wm * 64 + mi * 16 + r + 8 * h;
      const int s = inv[(size_t)be * C_ + mT * 128 + m];
      if (s >= 0) {
        const float sc = maxp[b * S_ + s];
#pragma unroll
        for (int ni = 0; ni < 2; ++ni) {
          const int n = wn * 32 + ni * 16 + lm;
          out[((size_t)b * S_ + s) * D_ + nT * 128 + n] = sc * acc[mi][ni][r];
        }
      }
    }
}

// ---------------------------------------------------------------------------
extern "C" void kernel_launch(void* const* d_in, const int* in_sizes, int n_in,
                              void* d_out, int out_size, void* d_ws,
                              size_t ws_size, hipStream_t stream) {
  const float* hs = (const float*)d_in[0];  // [B,S,D]
  const float* gw = (const float*)d_in[1];  // [D,E]
  const float* w1 = (const float*)d_in[2];  // [E,D,FF]
  const float* w2 = (const float*)d_in[3];  // [E,FF,D]
  const float* w3 = (const float*)d_in[4];  // [E,D,FF]

  float* out = (float*)d_out;                  // [B,S,D]
  float* logits = out + (size_t)B_ * S_ * D_;  // [B,S,E]

  char* ws = (char*)d_ws;
  int* sel = (int*)ws;                               // B*S ints
  float* maxp = (float*)(ws + (size_t)B_ * S_ * 4);  // B*S floats
  int* inv = (int*)(ws + (size_t)B_ * S_ * 8);       // B*E*C ints
  __bf16* X = (__bf16*)(ws + (size_t)B_ * S_ * 8 +
                        (size_t)B_ * E_ * C_ * 4);   // B*E*C*D bf16
  __bf16* H = X + (size_t)B_ * E_ * C_ * D_;         // B*E*C*FF bf16

  router_kernel<<<(B_ * S_) / 8, 256, 0, stream>>>(hs, gw, logits, sel, maxp);
  inv_init_kernel<<<(B_ * E_ * C_) / 256, 256, 0, stream>>>(inv);
  priority_kernel<<<B_, 256, 0, stream>>>(sel, inv);
  dispatch_kernel<<<B_ * E_ * C_, 256, 0, stream>>>(hs, inv, X);
  moe_gemm1_kernel<<<B_ * E_ * (C_ / 128) * (FF_ / 128), 256, 0, stream>>>(
      X, w1, w3, H);
  fill_kernel<<<(B_ * S_ * D_) / (256 * 4), 256, 0, stream>>>(hs, maxp, out);
  moe_gemm2_kernel<<<B_ * E_ * (C_ / 128) * (D_ / 128), 256, 0, stream>>>(
      H, w2, inv, maxp, out);
}
